// MRR_38087769981020
// MI455X (gfx1250) — matrix operand order repaired and checked by reading, not compiled, not run on hardware
//
#include <hip/hip_runtime.h>

// ---------------------------------------------------------------------------
// MRR (faithful to reference's positional argsort indexing) for MI455X gfx1250
//   1) yprep_kernel   : normalize queries, split into bf16 hi/lo -> ws
//   2) sim_gemm_kernel: bf16-split WMMA GEMM (cosine sims), async-to-LDS
//                       tile loads, coalesced staged stores -> ws (f32 [B][N])
//   3) select_kernel  : per-query radix-select of item at sorted position p
//   4) mrr_kernel     : mean(1 / (selected_index + 1))
// Workspace: B*N*4 (sims) + 256 (ranks pad) + 2*B*D*2 (Y hi/lo) bytes.
// ---------------------------------------------------------------------------

typedef __attribute__((ext_vector_type(16))) __bf16 v16bf;
typedef __attribute__((ext_vector_type(8)))  float  v8f;

#define BQ     64      // number of queries
#define DIM    128     // feature dimension
#define TILE_N 128     // product rows per workgroup
#define PPAD   132     // padded f32 row stride for P tile in LDS
#define YPAD   136     // padded bf16 row stride for Y hi/lo in LDS
#define EPS_N  1e-8f

// Ascending 32-bit key over DESCENDING similarity, order-preserving.
__device__ __forceinline__ unsigned ordKeyDesc(float s) {
    unsigned uf = __float_as_uint(-s);
    return (uf & 0x80000000u) ? ~uf : (uf | 0x80000000u);
}

// CDNA5 async copy: 16B global -> LDS, tracked by ASYNCcnt (no VGPR staging).
// LDS address operand = low 32 bits of the generic shared pointer
// (flat-shared addresses carry the wave-relative LDS offset in addr[31:0]).
__device__ __forceinline__ void async_copy_b128(void* lds_dst, const void* gsrc) {
    unsigned lds_off = (unsigned)(unsigned long long)lds_dst;
    asm volatile("global_load_async_to_lds_b128 %0, %1, off"
                 :: "v"(lds_off), "v"(gsrc) : "memory");
}
__device__ __forceinline__ void wait_asynccnt0() {
    asm volatile("s_wait_asynccnt 0x0" ::: "memory");
}

// ---------------------------------------------------------------------------
// Kernel 0: normalize queries once; emit pre-scaled bf16 hi/lo splits.
//   grid = BQ blocks x DIM threads.
// ---------------------------------------------------------------------------
__global__ __launch_bounds__(DIM) void yprep_kernel(
    const float* __restrict__ Y, __bf16* __restrict__ Yhi, __bf16* __restrict__ Ylo)
{
    __shared__ float red[DIM];
    const int b = blockIdx.x, t = threadIdx.x;
    const float v = Y[b * DIM + t];
    red[t] = v * v;
    __syncthreads();
    #pragma unroll
    for (int s = DIM / 2; s > 0; s >>= 1) {
        if (t < s) red[t] += red[t + s];
        __syncthreads();
    }
    const float r  = 1.0f / fmaxf(sqrtf(red[0]), EPS_N);
    const float vn = v * r;
    const __bf16 h = (__bf16)vn;
    Yhi[b * DIM + t] = h;
    Ylo[b * DIM + t] = (__bf16)(vn - (float)h);
}

// ---------------------------------------------------------------------------
// Kernel 1: cosine-similarity GEMM via bf16-split WMMA.
//   grid = ceil(N/128) blocks x 256 threads (8 waves).
// ---------------------------------------------------------------------------
__global__ __launch_bounds__(256) void sim_gemm_kernel(
    const __bf16* __restrict__ Yhig, const __bf16* __restrict__ Ylog,
    const float* __restrict__ P, float* __restrict__ sims, int N)
{
    extern __shared__ char smem[];
    float*  Plds = (float*)smem;                        // TILE_N * PPAD f32
    __bf16* Yhi  = (__bf16*)(Plds + TILE_N * PPAD);     // BQ * YPAD bf16
    __bf16* Ylo  = Yhi + BQ * YPAD;                     // BQ * YPAD bf16
    float*  prcp = (float*)(Ylo + BQ * YPAD);           // TILE_N f32
    float*  part = prcp + TILE_N;                       // 256 partials

    const int tid   = threadIdx.x;
    const int tile0 = blockIdx.x * TILE_N;

    // --- Async-copy Y hi/lo (2 x 16KB) into LDS: 8 x B128 chunks/thread ----
    {
        #pragma unroll
        for (int i = 0; i < 8; ++i) {
            int c   = tid + i * 256;          // 0..2047
            int arr = c >> 10;                // 0=hi, 1=lo
            int rc  = c & 1023;
            int row = rc >> 4;
            int k16 = rc & 15;                // 16B chunk within row
            __bf16*       dst = (arr ? Ylo : Yhi) + row * YPAD + k16 * 8;
            const __bf16* src = (arr ? Ylog : Yhig) + row * DIM + k16 * 8;
            async_copy_b128(dst, src);
        }
    }

    // --- Async-copy P tile (zero-fill OOB rows with ds stores) -------------
    {
        const int row = tid >> 1;             // 0..127
        const int c0  = (tid & 1) * 64;       // half-row
        const int j   = tile0 + row;
        float* dst = Plds + row * PPAD + c0;
        if (j < N) {
            const float* src = P + (size_t)j * DIM + c0;
            #pragma unroll
            for (int i = 0; i < 16; ++i)
                async_copy_b128(dst + i * 4, src + i * 4);
        } else {
            #pragma unroll
            for (int i = 0; i < 64; ++i) dst[i] = 0.0f;
        }
    }
    wait_asynccnt0();
    __syncthreads();

    // --- Reciprocal norms of P tile rows (2 threads per row) ---------------
    {
        const int row = tid >> 1;
        const int c0  = (tid & 1) * 64;
        const float* pr = Plds + row * PPAD + c0;
        float s = 0.0f;
        #pragma unroll
        for (int k = 0; k < 64; ++k) { float v = pr[k]; s += v * v; }
        part[tid] = s;
    }
    __syncthreads();
    if (tid < TILE_N)
        prcp[tid] = 1.0f / fmaxf(sqrtf(part[2 * tid] + part[2 * tid + 1]), EPS_N);
    __syncthreads();

    // --- WMMA compute -------------------------------------------------------
    const int wave = tid >> 5;          // 0..7: 16-row N-subtile
    const int lane = tid & 31;
    const int half = lane >> 4;
    const int l16  = lane & 15;
    const int rowB = wave * 16 + l16;
    const float pr = prcp[rowB];

    v8f acc[4] = {};

    #pragma unroll
    for (int kt = 0; kt < 4; ++kt) {
        // B fragment (32x16): lane = col N=rowB, elem e <-> K = kt*32+16*half+e
        v16bf bhi, blo;
        const float* prow = Plds + rowB * PPAD + kt * 32 + half * 16;
        #pragma unroll
        for (int e = 0; e < 16; ++e) {
            float v  = prow[e] * pr;
            __bf16 h = (__bf16)v;
            bhi[e] = h;
            blo[e] = (__bf16)(v - (float)h);
        }
        #pragma unroll
        for (int m = 0; m < 4; ++m) {
            // A fragment (16x32): elem e <-> K=kt*32+16*half+e
            v16bf ahi, alo;
            const int rowA = m * 16 + l16;
            const __bf16* yh = Yhi + rowA * YPAD + kt * 32 + half * 16;
            const __bf16* yl = Ylo + rowA * YPAD + kt * 32 + half * 16;
            #pragma unroll
            for (int e = 0; e < 8; ++e) {
                ahi[e]     = yh[e];      alo[e]     = yl[e];
                ahi[e + 8] = yh[e + 8]; alo[e + 8] = yl[e + 8];
            }
            acc[m] = __builtin_amdgcn_wmma_f32_16x16x32_bf16(
                false, ahi, false, bhi, (short)0, acc[m], false, false);
            acc[m] = __builtin_amdgcn_wmma_f32_16x16x32_bf16(
                false, ahi, false, blo, (short)0, acc[m], false, false);
            acc[m] = __builtin_amdgcn_wmma_f32_16x16x32_bf16(
                false, alo, false, bhi, (short)0, acc[m], false, false);
        }
    }

    // --- Stage results in LDS (reuse P tile region), then coalesced stores --
    __syncthreads();                      // all Plds reads done
    float* stage = Plds;                  // [BQ][PPAD] f32 (33KB < 66KB)
    #pragma unroll
    for (int m = 0; m < 4; ++m)
        #pragma unroll
        for (int r = 0; r < 8; ++r)
            stage[(m * 16 + half * 8 + r) * PPAD + wave * 16 + l16] = acc[m][r];
    __syncthreads();

    {
        const int b     = tid >> 2;             // 0..63
        const int cbase = (tid & 3) * 32;       // 0..96
        const bool n4ok = ((N & 3) == 0);
        #pragma unroll
        for (int i = 0; i < 8; ++i) {
            const int col = cbase + i * 4;
            const int j0  = tile0 + col;
            const float* sp = stage + b * PPAD + col;
            if (n4ok && (j0 + 4 <= N)) {
                float4 v = { sp[0], sp[1], sp[2], sp[3] };
                *(float4*)(sims + (size_t)b * N + j0) = v;
            } else {
                for (int e = 0; e < 4; ++e)
                    if (j0 + e < N) sims[(size_t)b * N + j0 + e] = sp[e];
            }
        }
    }
}

// ---------------------------------------------------------------------------
// Kernel 2: per-query radix selection. Finds the original index of the item
// at descending-sorted position q = gt_indices[b] (stable ties by index).
// One 256-thread workgroup per query; sims row lives in L2.
// ---------------------------------------------------------------------------
__global__ __launch_bounds__(256) void select_kernel(
    const float* __restrict__ sims, const int* __restrict__ gt,
    int* __restrict__ ranks, int N)
{
    __shared__ unsigned hist[256];
    __shared__ int cand[1024];
    __shared__ unsigned sh_prefix, sh_q, sh_cnt;

    const int b   = blockIdx.x;
    const int tid = threadIdx.x;
    const float* row = sims + (size_t)b * N;
    const bool vec4 = ((N & 3) == 0);
    const int  N4   = N >> 2;

    if (tid == 0) { sh_prefix = 0u; sh_q = (unsigned)gt[b]; }
    __syncthreads();

    for (int shift = 24; shift >= 0; shift -= 8) {
        hist[tid] = 0u;
        __syncthreads();
        const unsigned prefix = sh_prefix;
        const unsigned himask = (shift == 24) ? 0u : (0xFFFFFFFFu << (shift + 8));
        if (vec4) {
            const float4* row4 = (const float4*)row;
            for (int j = tid; j < N4; j += 256) {
                float4 v = row4[j];
                unsigned k0 = ordKeyDesc(v.x), k1 = ordKeyDesc(v.y);
                unsigned k2 = ordKeyDesc(v.z), k3 = ordKeyDesc(v.w);
                if (((k0 ^ prefix) & himask) == 0u) atomicAdd(&hist[(k0 >> shift) & 255u], 1u);
                if (((k1 ^ prefix) & himask) == 0u) atomicAdd(&hist[(k1 >> shift) & 255u], 1u);
                if (((k2 ^ prefix) & himask) == 0u) atomicAdd(&hist[(k2 >> shift) & 255u], 1u);
                if (((k3 ^ prefix) & himask) == 0u) atomicAdd(&hist[(k3 >> shift) & 255u], 1u);
            }
        } else {
            for (int j = tid; j < N; j += 256) {
                unsigned key = ordKeyDesc(row[j]);
                if (((key ^ prefix) & himask) == 0u)
                    atomicAdd(&hist[(key >> shift) & 255u], 1u);
            }
        }
        __syncthreads();
        if (tid == 0) {
            unsigned q = sh_q, p = sh_prefix;
            for (int c = 0; c < 256; ++c) {
                unsigned h = hist[c];
                if (q < h) { p |= ((unsigned)c) << shift; break; }
                q -= h;
            }
            sh_q = q; sh_prefix = p;
        }
        __syncthreads();
    }

    if (tid == 0) sh_cnt = 0u;
    __syncthreads();
    const unsigned target = sh_prefix;
    if (vec4) {
        const float4* row4 = (const float4*)row;
        for (int j = tid; j < N4; j += 256) {
            float4 v = row4[j];
            float vv[4] = { v.x, v.y, v.z, v.w };
            #pragma unroll
            for (int e = 0; e < 4; ++e)
                if (ordKeyDesc(vv[e]) == target) {
                    unsigned pos = atomicAdd(&sh_cnt, 1u);
                    if (pos < 1024u) cand[pos] = 4 * j + e;
                }
        }
    } else {
        for (int j = tid; j < N; j += 256) {
            if (ordKeyDesc(row[j]) == target) {
                unsigned pos = atomicAdd(&sh_cnt, 1u);
                if (pos < 1024u) cand[pos] = j;
            }
        }
    }
    __syncthreads();

    if (tid == 0) {
        int m = (sh_cnt < 1024u) ? (int)sh_cnt : 1024;
        int q = (int)sh_q;          // q-th smallest index among equal keys
        int last = -1;
        for (int t = 0; t <= q && t < m; ++t) {
            int cur = 0x7FFFFFFF;
            for (int i = 0; i < m; ++i) {
                int c = cand[i];
                if (c > last && c < cur) cur = c;
            }
            last = cur;
        }
        ranks[b] = last + 1;        // argsort(-sim)[b, gt[b]] + 1
    }
}

// ---------------------------------------------------------------------------
// Kernel 3: mrr = mean(1/rank)
// ---------------------------------------------------------------------------
__global__ void mrr_kernel(const int* __restrict__ ranks, float* __restrict__ out)
{
    __shared__ float s[BQ];
    int t = threadIdx.x;
    s[t] = 1.0f / (float)ranks[t];
    __syncthreads();
    if (t == 0) {
        float a = 0.0f;
        for (int i = 0; i < BQ; ++i) a += s[i];
        out[0] = a / (float)BQ;
    }
}

// ---------------------------------------------------------------------------
extern "C" void kernel_launch(void* const* d_in, const int* in_sizes, int n_in,
                              void* d_out, int out_size, void* d_ws, size_t ws_size,
                              hipStream_t stream)
{
    (void)n_in; (void)out_size; (void)ws_size;
    const float* Y  = (const float*)d_in[0];   // [B, D] f32
    const float* P  = (const float*)d_in[1];   // [N, D] f32
    const int*   gt = (const int*)d_in[2];     // [B] int32

    const int B = in_sizes[0] / DIM;           // 64
    const int N = in_sizes[1] / DIM;           // 500000

    // Workspace layout (16B-aligned sections):
    float*  sims  = (float*)d_ws;                                        // B*N f32
    char*   p     = (char*)d_ws + (size_t)B * N * sizeof(float);
    int*    ranks = (int*)p;                                             // B int
    __bf16* Yhig  = (__bf16*)(p + 256);                                  // B*D bf16
    __bf16* Ylog  = Yhig + (size_t)B * DIM;                              // B*D bf16

    yprep_kernel<<<B, DIM, 0, stream>>>(Y, Yhig, Ylog);

    const int nblk = (N + TILE_N - 1) / TILE_N;
    const size_t smem = (size_t)TILE_N * PPAD * sizeof(float)            // P tile / stage
                      + (size_t)2 * BQ * YPAD * sizeof(__bf16)           // Y hi/lo
                      + (size_t)TILE_N * sizeof(float)                   // prcp
                      + 256 * sizeof(float);                             // partials
    sim_gemm_kernel<<<nblk, 256, smem, stream>>>(Yhig, Ylog, P, sims, N);

    select_kernel<<<B, 256, 0, stream>>>(sims, gt, ranks, N);
    mrr_kernel<<<1, BQ, 0, stream>>>(ranks, (float*)d_out);
}
